// MODEL_44925357916709
// MI455X (gfx1250) — compile-verified, matrix operand order checked by reading, not run on hardware
//
// MI455X (gfx1250, CDNA5) implementation of the code2vec + Reformer-LSH reference.
//
// Roofline: ~70 GFLOP of GEMMs; gathered embedding tables (~36 MB) are L2-resident
// (192 MB L2), so the pipeline is compute-bound -> v_wmma_f32_16x16x32_f16
// (f16 inputs, f32 accum). Weights pre-converted once per launch to transposed
// f16 ([N][K]); activations flow as f16 (residual stream X stays f32), so both
// A and B fragments are contiguous 8-half runs matching the CDNA5 16-bit A/B
// VGPR striping (CDNA5 ISA 7.12.2) -> global_load_b128 clauses.
// Wave32: one 16x16 output tile per wave, 4 waves / 128-thread block.

#include <hip/hip_runtime.h>
#include <hip/hip_bf16.h>
#include <math.h>

typedef __attribute__((ext_vector_type(16))) _Float16 v16h;
typedef __attribute__((ext_vector_type(8)))  float    v8f;

#define WMMA_F16(a, b, c) \
  __builtin_amdgcn_wmma_f32_16x16x32_f16(false, (a), false, (b), (short)0, (c), false, false)

// ---- model dims ----
static const int B_ = 8, T_ = 1024, NP_ = 50, NC_ = 125;
static const int D_ = 256, DH_ = 32, DEPTH_ = 2;
static const int BUCKET_ = 64, NH_ = 4;
static const int BH_ = 64;                   // B*HEADS
static const int ROWS_ = B_ * T_;            // 8192
static const int SEQ_ = NH_ * T_;            // 4096
static const int NCH_ = 64;                  // NH*NB chunks per bh

// ---------------------------------------------------------------------------
// Fragment loaders. CDNA5 16-bit A-matrix 16x32 layout:
//   lanes 0-15 hold row M=lane, K slots {0..7, 16..23}; lanes 16-31: {8..15, 24..31}.
// A fragment = two contiguous 8-element runs at +half*8 and +16+half*8.
// B (K x N) is the dual layout (lane&15 = column n, same K striping), hence
// transposed-f16 weights [n][k] give contiguous loads.
// ---------------------------------------------------------------------------
__device__ __forceinline__ v16h frag_f32(const float* p, int half) {
  v16h r;
  const float* p0 = p + half * 8;
  const float* p1 = p + 16 + half * 8;
#pragma unroll
  for (int i = 0; i < 8; ++i) { r[i] = (_Float16)p0[i]; r[i + 8] = (_Float16)p1[i]; }
  return r;
}
__device__ __forceinline__ v16h frag_f16(const _Float16* p, int half) {
  v16h r;
  const _Float16* p0 = p + half * 8;
  const _Float16* p1 = p + 16 + half * 8;
#pragma unroll
  for (int i = 0; i < 8; ++i) { r[i] = p0[i]; r[i + 8] = p1[i]; }
  return r;
}

__device__ __forceinline__ float gelu_tanh(float x) {
  float x3 = x * x * x;
  return 0.5f * x * (1.f + tanhf(0.7978845608028654f * (x + 0.044715f * x3)));
}

// ---------------------------------------------------------------------------
// Weight prep: f32 [K][N] -> f16 transposed [N][K]
// ---------------------------------------------------------------------------
__global__ void k_prep_wt(const float* __restrict__ W, _Float16* __restrict__ Wt,
                          int K, int N) {
  int idx = blockIdx.x * blockDim.x + threadIdx.x;
  if (idx >= K * N) return;
  int k = idx / N, n = idx % N;
  Wt[(size_t)n * K + k] = (_Float16)W[idx];
}

// ---------------------------------------------------------------------------
// x = concat(c_embed[126], code_vec[128] (filled later), cur_result[2])
// ---------------------------------------------------------------------------
__global__ void k_build_x(const float* __restrict__ ce, const float* __restrict__ cr,
                          float* __restrict__ X) {
  int idx = blockIdx.x * blockDim.x + threadIdx.x;
  if (idx >= ROWS_ * 128) return;
  int row = idx >> 7, j = idx & 127;
  if (j < 126) X[(size_t)row * D_ + j] = ce[(size_t)row * 126 + j];
  else         X[(size_t)row * D_ + 254 + (j - 126)] = cr[(size_t)row * 2 + (j - 126)];
}

// ---------------------------------------------------------------------------
// Fused context encoder: per (b,t) block computes ctx = tanh(gather@cv_w)
// (50x384 @ 384x128 via WMMA), then attention-pools to code_vec -> X[:,126:254].
// Each 32-wide k-step lies entirely inside ONE embedding table (tables are 128
// wide, kk steps by 32), so table selection is wave-uniform and the A fragment
// is two contiguous vector loads -- no divergent scalar gathers. Rows >= NP
// use a clamped index (their ctx rows are never read by the reductions).
// ---------------------------------------------------------------------------
__global__ void __launch_bounds__(128)
k_ctx_codevec(const int* __restrict__ starts, const int* __restrict__ paths,
              const int* __restrict__ ends, const float* __restrict__ ne,
              const float* __restrict__ pe, const _Float16* __restrict__ cvwT,
              const float* __restrict__ avec, float* __restrict__ X) {
  __shared__ float ctxs[64][128];
  __shared__ float attw[64];
  const int bt = blockIdx.x;
  const int tid = threadIdx.x, lane = tid & 31, wave = tid >> 5;
  const int half = lane >> 4, l16 = lane & 15;
  const int tm = wave;
  const int row = tm * 16 + l16;
  const int rowc = (row < NP_) ? row : (NP_ - 1);   // clamp: rows >= NP unused
  const size_t g = (size_t)bt * NP_ + rowc;
  const int s0 = starts[g], p0 = paths[g], e0 = ends[g];
  const float* row0 = ne + (size_t)s0 * 128;        // k in [0,128)
  const float* row1 = pe + (size_t)p0 * 128;        // k in [128,256)
  const float* row2 = ne + (size_t)e0 * 128;        // k in [256,384)

  const v8f vzero = {0, 0, 0, 0, 0, 0, 0, 0};
  v8f acc[8];
#pragma unroll
  for (int nt = 0; nt < 8; ++nt) acc[nt] = vzero;

  for (int kk = 0; kk < 384; kk += 32) {
    const float* base = (kk < 128) ? (row0 + kk)
                      : (kk < 256) ? (row1 + kk - 128)
                                   : (row2 + kk - 256);
    v16h af = frag_f32(base, half);
#pragma unroll
    for (int nt = 0; nt < 8; ++nt) {
      v16h bf = frag_f16(cvwT + (size_t)(nt * 16 + l16) * 384 + kk, half);
      acc[nt] = WMMA_F16(af, bf, acc[nt]);
    }
  }
#pragma unroll
  for (int nt = 0; nt < 8; ++nt)
#pragma unroll
    for (int j = 0; j < 8; ++j)
      ctxs[tm * 16 + half * 8 + j][nt * 16 + l16] = tanhf(acc[nt][j]);
  __syncthreads();

  if (tid < NP_) {
    float s = 0.f;
    for (int c = 0; c < 128; ++c) s += ctxs[tid][c] * avec[c];
    attw[tid] = s;
  }
  __syncthreads();
  float cv = 0.f;
  for (int p = 0; p < NP_; ++p) cv += ctxs[p][tid] * attw[p];
  X[(size_t)bt * D_ + 126 + tid] = cv;
}

// ---------------------------------------------------------------------------
// LayerNorm over D=256 -> f16 output (feeds WMMA A-matrix directly).
// ---------------------------------------------------------------------------
__global__ void k_layernorm256(const float* __restrict__ x, const float* __restrict__ g,
                               const float* __restrict__ b, _Float16* __restrict__ y) {
  __shared__ float red[256];
  const int row = blockIdx.x, c = threadIdx.x;
  float v = x[(size_t)row * D_ + c];
  red[c] = v; __syncthreads();
  for (int s = 128; s > 0; s >>= 1) { if (c < s) red[c] += red[c + s]; __syncthreads(); }
  float mean = red[0] / 256.f; __syncthreads();
  float d = v - mean;
  red[c] = d * d; __syncthreads();
  for (int s = 128; s > 0; s >>= 1) { if (c < s) red[c] += red[c + s]; __syncthreads(); }
  float var = red[0] / 256.f;
  y[(size_t)row * D_ + c] = (_Float16)(d * rsqrtf(var + 1e-5f) * g[c] + b[c]);
}

// ---------------------------------------------------------------------------
// Generic WMMA GEMM: A[M,K] f16 @ Wt[N][K] f16 -> C f32 (optional) and/or
// Ch f16 (optional). One 16x16 tile per wave, 4 waves per block.
// epi: 0 none, 1 +resid, 2 +bias,gelu, 3 +bias, 4 +bias,+resid.
// ---------------------------------------------------------------------------
__global__ void __launch_bounds__(128)
k_gemm(const _Float16* __restrict__ A, const _Float16* __restrict__ Wt,
       float* __restrict__ C, _Float16* __restrict__ Ch,
       const float* __restrict__ resid, const float* __restrict__ bias,
       int M, int N, int K, int epi) {
  const int lane = threadIdx.x & 31, wave = threadIdx.x >> 5;
  const int ntn = (N + 15) >> 4;
  const int tiles = (M >> 4) * ntn;
  const int tile = blockIdx.x * 4 + wave;
  if (tile >= tiles) return;                 // wave-uniform: EXEC all-1s for WMMA
  const int tm = tile / ntn, tn = tile % ntn;
  const int half = lane >> 4, l16 = lane & 15;
  const int row = tm * 16 + l16;
  const int col = tn * 16 + l16;
  const int bcol = (col < N) ? col : 0;
  const _Float16* ap = A + (size_t)row * K;
  const _Float16* bp = Wt + (size_t)bcol * K;

  v8f acc = {0, 0, 0, 0, 0, 0, 0, 0};
  for (int kk = 0; kk < K; kk += 32) {
    if (kk + 32 < K) {                        // gfx1250 global_prefetch path
      __builtin_prefetch((const void*)(bp + kk + 32), 0, 3);
    }
    v16h af = frag_f16(ap + kk, half);
    v16h bf = frag_f16(bp + kk, half);
    acc = WMMA_F16(af, bf, acc);
  }
  if (col < N) {
    const int mbase = tm * 16 + half * 8;
#pragma unroll
    for (int j = 0; j < 8; ++j) {
      int m = mbase + j;
      float v = acc[j];
      if (epi >= 2) v += bias[col];
      if (epi == 2) v = gelu_tanh(v);
      if (epi == 1 || epi == 4) v += resid[(size_t)m * N + col];
      if (C)  C[(size_t)m * N + col] = v;
      if (Ch) Ch[(size_t)m * N + col] = (_Float16)v;
    }
  }
}

// ---------------------------------------------------------------------------
// LSH bucketing: bucket[bh][nh*T+t] = nh*16 + argmax_{j<16}(concat(r,-r)).
// ---------------------------------------------------------------------------
__global__ void k_lsh_buckets(const float* __restrict__ qk, const float* __restrict__ rot,
                              unsigned char* __restrict__ bkt) {
  int idx = blockIdx.x * blockDim.x + threadIdx.x;
  if (idx >= BH_ * T_) return;
  int bh = idx >> 10, t = idx & 1023;
  int b = bh >> 3, h = bh & 7;
  const float* q = qk + ((size_t)(b * T_ + t)) * D_ + h * DH_;
  float qr[DH_];
#pragma unroll
  for (int d = 0; d < DH_; ++d) qr[d] = q[d];
  for (int nh = 0; nh < NH_; ++nh) {
    float r8[8];
    for (int j = 0; j < 8; ++j) {
      float s = 0.f;
      for (int d = 0; d < DH_; ++d) s += qr[d] * rot[(d * NH_ + nh) * 8 + j];
      r8[j] = s;
    }
    float best = -3.0e38f; int bi = 0;
    for (int j = 0; j < 16; ++j) {                // argmax, first occurrence
      float v = (j < 8) ? r8[j] : -r8[j - 8];
      if (v > best) { best = v; bi = j; }
    }
    bkt[(size_t)bh * SEQ_ + nh * T_ + t] = (unsigned char)(nh * 16 + bi);
  }
}

// ---------------------------------------------------------------------------
// Stable counting sort by key = bucket*T + t (unique keys; buckets disjoint per
// hash round). Thread j owns bucket j; scans its round's positions in t order.
// ---------------------------------------------------------------------------
__global__ void k_lsh_sort(const unsigned char* __restrict__ bkt,
                           int* __restrict__ st, int* __restrict__ undo) {
  __shared__ int hist[64];
  __shared__ int offs[64];
  const int bh = blockIdx.x, tid = threadIdx.x;
  const unsigned char* bb = bkt + (size_t)bh * SEQ_;
  hist[tid] = 0;
  __syncthreads();
  for (int s = tid; s < SEQ_; s += 64) atomicAdd(&hist[bb[s]], 1);
  __syncthreads();
  if (tid == 0) {
    int run = 0;
    for (int j = 0; j < 64; ++j) { offs[j] = run; run += hist[j]; }
  }
  __syncthreads();
  const int j = tid, h = j >> 4;
  int base = offs[j];
  int* stb = st + (size_t)bh * SEQ_;
  int* ub  = undo + (size_t)bh * SEQ_;
  for (int t = 0; t < T_; ++t) {
    int s = h * T_ + t;
    if (bb[s] == j) { stb[base] = t; ub[s] = base; ++base; }
  }
}

// ---------------------------------------------------------------------------
// Chunked LSH attention: one (bh, chunk) per 128-thread block.
// bq 64x32; keys = this chunk + previous chunk (lookback), L2-normalized.
// dots via WMMA (4x8 tiles, K=32), mask, row LSE-softmax, P@V via WMMA
// (4x2 tiles, K=128). Outputs stay in sorted order (bo, lse).
// ---------------------------------------------------------------------------
__global__ void __launch_bounds__(128)
k_lsh_attn(const float* __restrict__ qk, const float* __restrict__ vv,
           const int* __restrict__ st, float* __restrict__ bo,
           float* __restrict__ lseg) {
  __shared__ _Float16 aq[64][32];
  __shared__ _Float16 bk[128][32];
  __shared__ _Float16 bvT[32][128];
  __shared__ float dots[64][128];
  __shared__ int qt[64];
  __shared__ int kt[128];
  const int c = blockIdx.x, bh = blockIdx.y;
  const int b = bh >> 3, h = bh & 7;
  const int tid = threadIdx.x, lane = tid & 31, wave = tid >> 5;
  const int* stb = st + (size_t)bh * SEQ_;

  if (tid < 64) {
    int t = stb[c * BUCKET_ + tid];
    qt[tid] = t;
    const float* q = qk + ((size_t)(b * T_ + t)) * D_ + h * DH_;
#pragma unroll
    for (int d = 0; d < DH_; ++d) aq[tid][d] = (_Float16)q[d];
  }
  {
    const int j = tid;
    const int cc = (j < 64) ? c : ((c + NCH_ - 1) & (NCH_ - 1));  // lookback chunk
    const int r = j & 63;
    const int t = stb[cc * BUCKET_ + r];
    kt[j] = t;
    const float* q = qk + ((size_t)(b * T_ + t)) * D_ + h * DH_;
    float tmp[DH_]; float ss = 0.f;
#pragma unroll
    for (int d = 0; d < DH_; ++d) { tmp[d] = q[d]; ss += tmp[d] * tmp[d]; }
    const float inv = 1.f / (sqrtf(ss) + 1e-12f);
#pragma unroll
    for (int d = 0; d < DH_; ++d) bk[j][d] = (_Float16)(tmp[d] * inv);
    const float* vp = vv + ((size_t)(b * T_ + t)) * D_ + h * DH_;
#pragma unroll
    for (int d = 0; d < DH_; ++d) bvT[d][j] = (_Float16)vp[d];
  }
  __syncthreads();

  const int half = lane >> 4, l16 = lane & 15;
  const int tm = wave;
  const v8f vzero = {0, 0, 0, 0, 0, 0, 0, 0};
#pragma unroll
  for (int tn = 0; tn < 8; ++tn) {
    v16h af = frag_f16(&aq[tm * 16 + l16][0], half);
    v16h bf = frag_f16(&bk[tn * 16 + l16][0], half);
    v8f acc = WMMA_F16(af, bf, vzero);
#pragma unroll
    for (int j = 0; j < 8; ++j) {
      int m = tm * 16 + half * 8 + j;
      int n = tn * 16 + l16;
      float dv = acc[j] * 0.17677669529663687f;     // DH^-0.5
      int qtt = qt[m], ktt = kt[n];
      dv = (qtt < ktt) ? -1.0e9f : ((qtt == ktt) ? -5.0e4f : dv);
      dots[m][n] = dv;
    }
  }
  __syncthreads();

  if (tid < 64) {
    float mx = -3.0e38f;
    for (int n = 0; n < 128; ++n) mx = fmaxf(mx, dots[tid][n]);
    float s = 0.f;
    for (int n = 0; n < 128; ++n) s += expf(dots[tid][n] - mx);
    float l = logf(s) + mx;
    for (int n = 0; n < 128; ++n) dots[tid][n] = expf(dots[tid][n] - l);
    lseg[(size_t)bh * SEQ_ + c * BUCKET_ + tid] = l;
  }
  __syncthreads();

#pragma unroll
  for (int tn = 0; tn < 2; ++tn) {
    v8f acc = vzero;
#pragma unroll
    for (int ks = 0; ks < 4; ++ks) {
      v16h af = frag_f32(&dots[tm * 16 + l16][ks * 32], half);
      v16h bf = frag_f16(&bvT[tn * 16 + l16][ks * 32], half);
      acc = WMMA_F16(af, bf, acc);
    }
#pragma unroll
    for (int j = 0; j < 8; ++j) {
      int m = tm * 16 + half * 8 + j;
      int n = tn * 16 + l16;
      bo[((size_t)bh * SEQ_ + c * BUCKET_ + m) * DH_ + n] = acc[j];
    }
  }
}

// ---------------------------------------------------------------------------
// Unsort + hash-round softmax combine -> O f16 in (B,T,D) layout (feeds Wout).
// ---------------------------------------------------------------------------
__global__ void k_lsh_combine(const float* __restrict__ bo, const float* __restrict__ lseg,
                              const int* __restrict__ undo, _Float16* __restrict__ O) {
  int idx = blockIdx.x * blockDim.x + threadIdx.x;
  if (idx >= BH_ * T_) return;
  int bh = idx >> 10, t = idx & 1023;
  int b = bh >> 3, h = bh & 7;
  const int* ub = undo + (size_t)bh * SEQ_;
  int p[NH_]; float l[NH_];
  float mx = -3.0e38f;
#pragma unroll
  for (int hh = 0; hh < NH_; ++hh) {
    p[hh] = ub[hh * T_ + t];
    l[hh] = lseg[(size_t)bh * SEQ_ + p[hh]];
    mx = fmaxf(mx, l[hh]);
  }
  float w[NH_]; float wsum = 0.f;
#pragma unroll
  for (int hh = 0; hh < NH_; ++hh) { w[hh] = expf(l[hh] - mx); wsum += w[hh]; }
  float o[DH_];
#pragma unroll
  for (int d = 0; d < DH_; ++d) o[d] = 0.f;
#pragma unroll
  for (int hh = 0; hh < NH_; ++hh) {
    const float ww = w[hh] / wsum;
    const float* br = bo + ((size_t)bh * SEQ_ + p[hh]) * DH_;
    for (int d = 0; d < DH_; ++d) o[d] += ww * br[d];
  }
  _Float16* op = O + ((size_t)(b * T_ + t)) * D_ + h * DH_;
  for (int d = 0; d < DH_; ++d) op[d] = (_Float16)o[d];
}

// ---------------------------------------------------------------------------
// Head: masked pooled logit + BCE, atomic reduction.
// ---------------------------------------------------------------------------
__global__ void k_head_loss(const float* __restrict__ pred, const float* __restrict__ tc,
                            const float* __restrict__ result, float* __restrict__ fpb,
                            float* __restrict__ red) {
  int i = blockIdx.x * blockDim.x + threadIdx.x;
  if (i >= ROWS_) return;
  float p1 = 0.f, nc = 0.f;
  for (int c = 0; c < NC_ + 1; ++c) {
    float t = tc[(size_t)i * (NC_ + 1) + c];
    p1 += pred[(size_t)i * (NC_ + 1) + c] * t;
    nc += t;
  }
  float m = (nc > 0.f) ? 1.f : 0.f;
  float fp = p1 / ((nc > 0.f) ? nc : 1.f);
  float tg = result[i];
  float bce = fmaxf(fp, 0.f) - fp * tg + log1pf(expf(-fabsf(fp)));
  fpb[i] = fp;
  fpb[ROWS_ + i] = m;
  atomicAdd(&red[0], bce * m);
  atomicAdd(&red[1], m);
}

__global__ void k_finalize(const float* __restrict__ fpb, const float* __restrict__ red,
                           const float* __restrict__ result, float* __restrict__ out) {
  int i = blockIdx.x * blockDim.x + threadIdx.x;
  if (i == 0) out[0] = red[0] / red[1];
  if (i < ROWS_) {
    float fp = fpb[i], m = fpb[ROWS_ + i];
    out[1 + i] = (1.f / (1.f + expf(-fp))) * m;
    out[1 + ROWS_ + i] = result[i] * m;
  }
}

// ---------------------------------------------------------------------------
// Host orchestration
// ---------------------------------------------------------------------------
extern "C" void kernel_launch(void* const* d_in, const int* in_sizes, int n_in,
                              void* d_out, int out_size, void* d_ws, size_t ws_size,
                              hipStream_t stream) {
  (void)in_sizes; (void)n_in; (void)out_size; (void)ws_size;
  // setup_inputs() dict order (params dict flattened in insertion order)
  const int*   starts    = (const int*)d_in[2];
  const int*   paths     = (const int*)d_in[3];
  const int*   ends      = (const int*)d_in[4];
  const float* target_c  = (const float*)d_in[6];
  const float* result    = (const float*)d_in[7];
  const float* c_embed   = (const float*)d_in[8];
  const float* cur_res   = (const float*)d_in[9];
  const float* node_emb  = (const float*)d_in[10];
  const float* path_emb  = (const float*)d_in[11];
  const float* cv_w      = (const float*)d_in[12];
  const float* avec      = (const float*)d_in[13];
  const float* Wp        = (const float*)d_in[14];
  const float* bp        = (const float*)d_in[15];
  const float* rotations = (const float*)d_in[38];
  // per-layer params at 16 + l*11: ln1_g, ln1_b, Wqk, Wv, Wout, ln2_g, ln2_b, W1, b1, W2, b2
  auto LP = [&](int l, int k) { return (const float*)d_in[16 + l * 11 + k]; };

  // ---- workspace layout ----
  size_t off = 0;
  auto alloc = [&](size_t bytes) {
    void* p = (char*)d_ws + off;
    off += (bytes + 255) & ~(size_t)255;
    return p;
  };
  float*     X    = (float*)alloc((size_t)ROWS_ * D_ * 4);       // f32 residual stream
  _Float16*  Hh   = (_Float16*)alloc((size_t)ROWS_ * D_ * 2);    // LN output (f16)
  float*     QK   = (float*)alloc((size_t)ROWS_ * D_ * 4);
  float*     VV   = (float*)alloc((size_t)ROWS_ * D_ * 4);
  _Float16*  OOh  = (_Float16*)alloc((size_t)ROWS_ * D_ * 2);    // attn out (f16)
  _Float16*  Xh   = (_Float16*)alloc((size_t)ROWS_ * D_ * 2);    // final x (f16, head A)
  float*     PRED = (float*)alloc((size_t)ROWS_ * (NC_ + 1) * 4);
  float*     BO   = (float*)alloc((size_t)BH_ * SEQ_ * DH_ * 4); // 32MB
  _Float16*  FFNh = (_Float16*)BO;   // reuse: attention done before FFN each layer
  float*     LSE  = (float*)alloc((size_t)BH_ * SEQ_ * 4);
  int*       ST   = (int*)alloc((size_t)BH_ * SEQ_ * 4);
  int*       UNDO = (int*)alloc((size_t)BH_ * SEQ_ * 4);
  unsigned char* BKT = (unsigned char*)alloc((size_t)BH_ * SEQ_);
  float*     FPB  = (float*)alloc((size_t)2 * ROWS_ * 4);
  float*     RED  = (float*)alloc(256);
  // transposed f16 weights
  size_t ho = 0;
  _Float16* WTBASE = (_Float16*)alloc((size_t)2000000 * 2);
  auto halloc = [&](size_t n) { _Float16* p = WTBASE + ho; ho += n; return p; };
  _Float16* CVWT = halloc(384 * 128);
  _Float16 *WQKT[2], *WVT[2], *WOUTT[2], *W1T[2], *W2T[2];
  for (int l = 0; l < DEPTH_; ++l) {
    WQKT[l]  = halloc(256 * 256);
    WVT[l]   = halloc(256 * 256);
    WOUTT[l] = halloc(256 * 256);
    W1T[l]   = halloc(256 * 1024);
    W2T[l]   = halloc(1024 * 256);
  }
  _Float16* WPT = halloc(126 * 256);

  auto prep = [&](const float* W, _Float16* Wt, int K, int N) {
    int n = K * N;
    k_prep_wt<<<(n + 255) / 256, 256, 0, stream>>>(W, Wt, K, N);
  };
  prep(cv_w, CVWT, 384, 128);
  for (int l = 0; l < DEPTH_; ++l) {
    prep(LP(l, 2), WQKT[l], 256, 256);
    prep(LP(l, 3), WVT[l], 256, 256);
    prep(LP(l, 4), WOUTT[l], 256, 256);
    prep(LP(l, 7), W1T[l], 256, 1024);
    prep(LP(l, 9), W2T[l], 1024, 256);
  }
  prep(Wp, WPT, 256, 126);

  auto gemm = [&](const _Float16* A, const _Float16* Wt, float* C, _Float16* Ch,
                  const float* resid, const float* bias, int M, int N, int K, int epi) {
    int tiles = (M / 16) * ((N + 15) / 16);
    k_gemm<<<(tiles + 3) / 4, 128, 0, stream>>>(A, Wt, C, Ch, resid, bias, M, N, K, epi);
  };

  // build x
  k_build_x<<<(ROWS_ * 128 + 255) / 256, 256, 0, stream>>>(c_embed, cur_res, X);
  k_ctx_codevec<<<ROWS_, 128, 0, stream>>>(starts, paths, ends, node_emb, path_emb,
                                           CVWT, avec, X);

  for (int l = 0; l < DEPTH_; ++l) {
    const float* rot = rotations + (size_t)l * DH_ * NH_ * 8;
    k_layernorm256<<<ROWS_, 256, 0, stream>>>(X, LP(l, 0), LP(l, 1), Hh);
    gemm(Hh, WQKT[l], QK, nullptr, nullptr, nullptr, ROWS_, 256, 256, 0);
    gemm(Hh, WVT[l], VV, nullptr, nullptr, nullptr, ROWS_, 256, 256, 0);
    k_lsh_buckets<<<(BH_ * T_ + 255) / 256, 256, 0, stream>>>(QK, rot, BKT);
    k_lsh_sort<<<BH_, 64, 0, stream>>>(BKT, ST, UNDO);
    k_lsh_attn<<<dim3(NCH_, BH_), 128, 0, stream>>>(QK, VV, ST, BO, LSE);
    k_lsh_combine<<<(BH_ * T_ + 127) / 128, 128, 0, stream>>>(BO, LSE, UNDO, OOh);
    gemm(OOh, WOUTT[l], X, nullptr, X, nullptr, ROWS_, 256, 256, 1);      // x += o@Wout
    k_layernorm256<<<ROWS_, 256, 0, stream>>>(X, LP(l, 5), LP(l, 6), Hh);
    gemm(Hh, W1T[l], nullptr, FFNh, nullptr, LP(l, 8), ROWS_, 1024, 256, 2); // gelu
    gemm(FFNh, W2T[l], X, (l == DEPTH_ - 1) ? Xh : nullptr,
         X, LP(l, 10), ROWS_, 256, 1024, 4);                              // x += .@W2+b2
  }

  gemm(Xh, WPT, PRED, nullptr, nullptr, bp, ROWS_, 126, 256, 3);          // pred
  hipMemsetAsync(RED, 0, 2 * sizeof(float), stream);
  k_head_loss<<<(ROWS_ + 255) / 256, 256, 0, stream>>>(PRED, target_c, result, FPB, RED);
  k_finalize<<<(ROWS_ + 255) / 256, 256, 0, stream>>>(FPB, RED, result, (float*)d_out);
}
